// GCNII_60902636257284
// MI455X (gfx1250) — matrix operand order repaired and checked by reading, not compiled, not run on hardware
//
#include <hip/hip_runtime.h>
#include <hip/hip_bf16.h>
#include <cmath>

// ---------------------------------------------------------------------------
// GCNII inference for MI455X (gfx1250, wave32, WMMA bf16 -> f32 accum)
// ---------------------------------------------------------------------------

typedef __attribute__((ext_vector_type(16))) __bf16 v16bf;
typedef __attribute__((ext_vector_type(8)))  float  v8f;
typedef int v4i_vs __attribute__((vector_size(16)));   // matches builtin param type

union Frag16 {            // 16 bf16 = 32 bytes = two uint4
    uint4  q[2];
    v16bf  v;
};

// gfx1250 async global->LDS copy (ASYNCcnt path), guarded by __has_builtin
#if defined(__gfx1250__) && __has_builtin(__builtin_amdgcn_global_load_async_to_lds_b128)
#define USE_ASYNC_LDS 1
#else
#define USE_ASYNC_LDS 0
#endif

__device__ __forceinline__ void async_copy16(const void* gsrc, void* ldst) {
#if USE_ASYNC_LDS
    __builtin_amdgcn_global_load_async_to_lds_b128(
        (__attribute__((address_space(1))) v4i_vs*)gsrc,
        (__attribute__((address_space(3))) v4i_vs*)ldst, 0, 0);
#else
    *(uint4*)ldst = *(const uint4*)gsrc;
#endif
}
__device__ __forceinline__ void async_wait_all() {
#if USE_ASYNC_LDS
#if __has_builtin(__builtin_amdgcn_s_wait_asynccnt)
    __builtin_amdgcn_s_wait_asynccnt(0);
#else
    asm volatile("s_wait_asynccnt 0x0" ::: "memory");
#endif
#endif
}

// round-to-nearest-even f32 -> bf16 (bit-level)
__device__ __forceinline__ unsigned int f2bf(float f) {
    unsigned int x = __float_as_uint(f);
    return (x + 0x7FFFu + ((x >> 16) & 1u)) >> 16;
}
__device__ __forceinline__ unsigned int pack_bf16x2(float lo, float hi) {
    return f2bf(lo) | (f2bf(hi) << 16);
}

// --- WMMA fragment loaders (ISA 7.12.2 layouts, wave32) ---------------------
// A (16x32 bf16): lane m = lane&15, half = lane>>4.
//   VGPR0-3 : K = kb + 8*half + 0..7   (contiguous -> one b128)
//   VGPR4-7 : K = kb + 16 + 8*half + 0..7
__device__ __forceinline__ v16bf load_a_frag(const unsigned short* lsA,
                                             int mt, int kb, int lane) {
    int m    = mt * 16 + (lane & 15);
    int half = lane >> 4;
    const unsigned short* p = lsA + m * 128 + kb + 8 * half;
    Frag16 f;
    f.q[0] = *(const uint4*)(p);
    f.q[1] = *(const uint4*)(p + 16);
    return f.v;
}
// B (32x16 bf16): lane holds column N = lane&15; K = kb + 8*half + 0..7, then kb + 16 + 8*half + 0..7 (row-major B).
template<int NPAD>
__device__ __forceinline__ v16bf load_b_frag(const unsigned short* lsB,
                                             int kb, int nt, int lane) {
    const unsigned short* p = lsB + (kb + 8 * (lane >> 4)) * NPAD + nt * 16 + (lane & 15);
    Frag16 f;
    unsigned short* e = (unsigned short*)&f;
    #pragma unroll
    for (int s = 0; s < 8; ++s) {
        e[s]     = p[s * NPAD];
        e[s + 8] = p[(s + 16) * NPAD];
    }
    return f.v;
}

// ---------------------------------------------------------------------------
// Weight conversion: f32 -> bf16, row-major [K][N].  W_out padded 40 -> 48.
// ---------------------------------------------------------------------------
__global__ __launch_bounds__(256)
void k_cvt_weights(const float* __restrict__ Win,  const float* __restrict__ conv,
                   const float* __restrict__ Wout, unsigned short* __restrict__ wq) {
    int i = blockIdx.x * 256 + threadIdx.x;
    if (i < 16384) {                               // W_in 128x128
        wq[i] = (unsigned short)f2bf(Win[i]);
    } else if (i < 16384 * 9) {                    // conv_w 8 x 128x128
        wq[i] = (unsigned short)f2bf(conv[i - 16384]);
    } else if (i < 16384 * 9 + 128 * 48) {         // W_out 128x40 padded to 48
        int j = i - 16384 * 9;
        int k = j / 48, n = j % 48;
        wq[i] = (n < 40) ? (unsigned short)f2bf(Wout[k * 40 + n]) : 0;
    }
}

// ---------------------------------------------------------------------------
// Generic GEMM:  D[M x Nout] = A(f32)[M x 128] @ B(bf16)[128 x NT*16] + bias
// Block = 256 threads (8 waves), tile 128(M) x NT*16(N), K=128 fully in LDS.
// ---------------------------------------------------------------------------
template<int NT>
__global__ __launch_bounds__(256)
void k_gemm(const float* __restrict__ A, const unsigned short* __restrict__ B,
            const float* __restrict__ bias, float* __restrict__ D,
            int M, int Nout, int ldD) {
    __shared__ __align__(16) unsigned short lsA[128 * 128];
    __shared__ __align__(16) unsigned short lsB[128 * 16 * NT];
    const int tid     = threadIdx.x;
    const int rowBase = blockIdx.x * 128;

    // stage B (already bf16): async global->LDS, overlapped with A conversion
    for (int i = tid; i < 128 * NT * 16 / 8; i += 256)
        async_copy16((const uint4*)B + i, (uint4*)lsB + i);
    // stage A: f32 -> bf16 (row-guarded)
    for (int i = tid; i < 128 * 32; i += 256) {
        int r = i >> 5, c4 = i & 31;
        int gr = rowBase + r;
        float4 v = make_float4(0.f, 0.f, 0.f, 0.f);
        if (gr < M) v = *(const float4*)(A + (size_t)gr * 128 + c4 * 4);
        ((unsigned int*)lsA)[r * 64 + c4 * 2]     = pack_bf16x2(v.x, v.y);
        ((unsigned int*)lsA)[r * 64 + c4 * 2 + 1] = pack_bf16x2(v.z, v.w);
    }
    async_wait_all();
    __syncthreads();

    const int wave = tid >> 5, lane = tid & 31;
    v8f acc[NT];
    #pragma unroll
    for (int nt = 0; nt < NT; ++nt) acc[nt] = (v8f)(0.f);

    #pragma unroll
    for (int kb = 0; kb < 128; kb += 32) {
        // preload all fragments for this K-step, then drain WMMAs back-to-back
        v16bf a = load_a_frag(lsA, wave, kb, lane);
        v16bf bfr[NT];
        #pragma unroll
        for (int nt = 0; nt < NT; ++nt)
            bfr[nt] = load_b_frag<NT * 16>(lsB, kb, nt, lane);
        #pragma unroll
        for (int nt = 0; nt < NT; ++nt)
            acc[nt] = __builtin_amdgcn_wmma_f32_16x16x32_bf16(
                false, a, false, bfr[nt], (short)0, acc[nt], false, false);
    }

    const int half = lane >> 4, nlo = lane & 15;
    #pragma unroll
    for (int nt = 0; nt < NT; ++nt) {
        #pragma unroll
        for (int r = 0; r < 8; ++r) {
            int m = rowBase + wave * 16 + 8 * half + r;
            int n = nt * 16 + nlo;
            if (m < M && n < Nout)
                D[(size_t)m * ldD + n] = acc[nt][r] + bias[n];
        }
    }
}

// ---------------------------------------------------------------------------
// Fused GCNII layer:
//   support = (1-a)*hi + a*h0
//   h       = relu( theta*(support @ W) + (1-theta)*support )
// ---------------------------------------------------------------------------
__global__ __launch_bounds__(256)
void k_layer(const float* __restrict__ hi, const float* __restrict__ h0,
             const unsigned short* __restrict__ B, float theta, float alpha,
             float* __restrict__ hout, int M) {
    __shared__ __align__(16) unsigned short lsA[128 * 128];
    __shared__ __align__(16) unsigned short lsB[128 * 128];
    const int tid     = threadIdx.x;
    const int rowBase = blockIdx.x * 128;

    for (int i = tid; i < 2048; i += 256)
        async_copy16((const uint4*)B + i, (uint4*)lsB + i);
    for (int i = tid; i < 128 * 32; i += 256) {
        int r = i >> 5, c4 = i & 31;
        int gr = rowBase + r;
        float4 s = make_float4(0.f, 0.f, 0.f, 0.f);
        if (gr < M) {
            float4 a = *(const float4*)(hi + (size_t)gr * 128 + c4 * 4);
            float4 b = *(const float4*)(h0 + (size_t)gr * 128 + c4 * 4);
            s.x = (1.f - alpha) * a.x + alpha * b.x;
            s.y = (1.f - alpha) * a.y + alpha * b.y;
            s.z = (1.f - alpha) * a.z + alpha * b.z;
            s.w = (1.f - alpha) * a.w + alpha * b.w;
        }
        ((unsigned int*)lsA)[r * 64 + c4 * 2]     = pack_bf16x2(s.x, s.y);
        ((unsigned int*)lsA)[r * 64 + c4 * 2 + 1] = pack_bf16x2(s.z, s.w);
    }
    async_wait_all();
    __syncthreads();

    const int wave = tid >> 5, lane = tid & 31;
    v8f acc[8];
    #pragma unroll
    for (int nt = 0; nt < 8; ++nt) acc[nt] = (v8f)(0.f);

    #pragma unroll
    for (int kb = 0; kb < 128; kb += 32) {
        v16bf a = load_a_frag(lsA, wave, kb, lane);
        v16bf bfr[8];
        #pragma unroll
        for (int nt = 0; nt < 8; ++nt)
            bfr[nt] = load_b_frag<128>(lsB, kb, nt, lane);
        #pragma unroll
        for (int nt = 0; nt < 8; ++nt)
            acc[nt] = __builtin_amdgcn_wmma_f32_16x16x32_bf16(
                false, a, false, bfr[nt], (short)0, acc[nt], false, false);
    }

    const int half = lane >> 4, nlo = lane & 15;
    #pragma unroll
    for (int nt = 0; nt < 8; ++nt) {
        #pragma unroll
        for (int r = 0; r < 8; ++r) {
            int m = rowBase + wave * 16 + 8 * half + r;
            int n = nt * 16 + nlo;
            if (m < M) {
                // recompute f32 support for the identity-mapping term (L2 hit)
                float s = (1.f - alpha) * hi[(size_t)m * 128 + n]
                        + alpha * h0[(size_t)m * 128 + n];
                float o = theta * acc[nt][r] + (1.f - theta) * s;
                hout[(size_t)m * 128 + n] = fmaxf(o, 0.f);
            }
        }
    }
}

// ---------------------------------------------------------------------------
// SpMM scatter: hi[dst] += val * h[src].  One wave per 4 edges, float4 per lane,
// hardware global_atomic_add_f32 into L2 (h fits entirely in 192MB L2).
// ---------------------------------------------------------------------------
__global__ __launch_bounds__(256)
void k_spmm(const int* __restrict__ src, const int* __restrict__ dst,
            const float* __restrict__ val, const float* __restrict__ h,
            float* __restrict__ acc, int E) {
    const int lane = threadIdx.x & 31;
    const int wave = blockIdx.x * 8 + (threadIdx.x >> 5);
    int e0 = wave * 4;
    #pragma unroll
    for (int i = 0; i < 4; ++i) {
        int e = e0 + i;
        if (e < E) {
            int   s = src[e];
            int   d = dst[e];
            float v = val[e];
            float4 hv = *(const float4*)(h + (size_t)s * 128 + lane * 4);
            float* p  = acc + (size_t)d * 128 + lane * 4;
            unsafeAtomicAdd(p + 0, v * hv.x);
            unsafeAtomicAdd(p + 1, v * hv.y);
            unsafeAtomicAdd(p + 2, v * hv.z);
            unsafeAtomicAdd(p + 3, v * hv.w);
        }
    }
}

// ---------------------------------------------------------------------------
extern "C" void kernel_launch(void* const* d_in, const int* in_sizes, int n_in,
                              void* d_out, int out_size, void* d_ws, size_t ws_size,
                              hipStream_t stream) {
    const float* x        = (const float*)d_in[0];
    const int*   edge_src = (const int*)  d_in[1];
    const int*   edge_dst = (const int*)  d_in[2];
    const float* edge_val = (const float*)d_in[3];
    const float* W_in     = (const float*)d_in[4];
    const float* b_in     = (const float*)d_in[5];
    const float* conv_w   = (const float*)d_in[6];
    const float* W_out    = (const float*)d_in[7];
    const float* b_out    = (const float*)d_in[8];

    const int Nn = in_sizes[0] / 128;      // 100000 nodes
    const int E  = in_sizes[1];            // 3.2M edges
    const size_t nh = (size_t)Nn * 128;

    // workspace layout: h0 | h | hi | bf16 weights
    float* h0 = (float*)d_ws;
    float* h  = h0 + nh;
    float* hi = h + nh;
    unsigned short* wq    = (unsigned short*)(hi + nh);
    unsigned short* wIn   = wq;                 // 128x128
    unsigned short* wConv = wq + 16384;         // 8 x 128x128
    unsigned short* wOut  = wq + 16384 * 9;     // 128x48 (padded)

    const int mblocks = (Nn + 127) / 128;
    const int cvt_elems = 16384 * 9 + 128 * 48;

    k_cvt_weights<<<(cvt_elems + 255) / 256, 256, 0, stream>>>(W_in, conv_w, W_out, wq);

    // fc_in:  h0 = x @ W_in + b_in
    k_gemm<8><<<mblocks, 256, 0, stream>>>(x, wIn, b_in, h0, Nn, 128, 128);

    for (int i = 0; i < 8; ++i) {
        float theta = logf(0.5f / (float)(i + 1) + 1.0f);
        (void)hipMemsetAsync(hi, 0, nh * sizeof(float), stream);
        k_spmm<<<(E + 31) / 32, 256, 0, stream>>>(
            edge_src, edge_dst, edge_val, (i == 0) ? h0 : h, hi, E);
        k_layer<<<mblocks, 256, 0, stream>>>(
            hi, h0, wConv + (size_t)i * 16384, theta, 0.1f, h, Nn);
    }

    // fc_out: out = h @ W_out + b_out   (Nout=40, padded B to 48)
    k_gemm<3><<<mblocks, 256, 0, stream>>>(h, wOut, b_out, (float*)d_out, Nn, 40, 40);
}
